// multiAttentionHead_58987080843315
// MI455X (gfx1250) — matrix operand order repaired and checked by reading, not compiled, not run on hardware
//
#include <hip/hip_runtime.h>
#include <hip/hip_bf16.h>

#define B_  4
#define S_  2048
#define D_  640
#define H_  10
#define KD_ 64

typedef __attribute__((ext_vector_type(16))) _Float16 v16h;
typedef __attribute__((ext_vector_type(8)))  _Float16 v8h;
typedef __attribute__((ext_vector_type(8)))  float    v8f;

// ---------------- WMMA fragment helpers (CDNA5 wave32 layouts) ----------------
// All loaders issue contiguous 16B accesses (-> ds_load_b128 / global_load_b128).

// A matrix 16x32 f16, row-major [M][K] source, lda multiple of 8 halfs.
// lanes 0-15: M=lane, K={0..7,16..23}; lanes 16-31: M=lane-16, K={8..15,24..31}
static __device__ inline v16h load_a16x32(const _Float16* __restrict__ src, int lda) {
  const int lane = threadIdx.x & 31;
  const int m  = lane & 15;
  const int kb = (lane >> 4) << 3;          // 0 or 8
  const _Float16* r = src + (size_t)m * lda + kb;
  v8h lo = *(const v8h*)(r);
  v8h hi = *(const v8h*)(r + 16);
  return __builtin_shufflevector(lo, hi, 0, 1, 2, 3, 4, 5, 6, 7,
                                 8, 9, 10, 11, 12, 13, 14, 15);
}

// B matrix 32x16 f16 given in COLUMN-major form: src is [N=16][K=32] row-major
// with leading dim ld (multiple of 8 halfs). Lane n reads two 8-half chunks.
// lanes 0-15: N=lane, K={0..7,16..23}; lanes 16-31: N=lane-16, K={8..15,24..31}
static __device__ inline v16h load_bT(const _Float16* __restrict__ src, int ld) {
  const int lane = threadIdx.x & 31;
  const int n  = lane & 15;
  const int kb = (lane >> 4) << 3;          // 0 or 8
  const _Float16* r = src + (size_t)n * ld + kb;
  v8h lo = *(const v8h*)(r);
  v8h hi = *(const v8h*)(r + 16);
  return __builtin_shufflevector(lo, hi, 0, 1, 2, 3, 4, 5, 6, 7,
                                 8, 9, 10, 11, 12, 13, 14, 15);
}

static __device__ inline v8f wmma16(v16h a, v16h b, v8f c) {
  return __builtin_amdgcn_wmma_f32_16x16x32_f16(false, a, false, b, (short)0, c,
                                                false, false);
}

// ---------------- kernel 0a: fp32 -> f16 straight conversion ----------------
__global__ void mha_cvt_kernel(const float* __restrict__ src,
                               _Float16* __restrict__ dst, int n) {
  int i = blockIdx.x * blockDim.x + threadIdx.x;
  int stride = gridDim.x * blockDim.x;
  for (; i < n; i += stride) dst[i] = (_Float16)src[i];
}

// ---------------- kernel 0b: fp32 [R][C] -> f16 transposed [C][R] ------------
// grid (C/32, R/32, batch), block 256. R, C multiples of 32.
__global__ __launch_bounds__(256) void mha_cvt_t_kernel(
    const float* __restrict__ src, _Float16* __restrict__ dst, int R, int C) {
  __shared__ float t[32][33];
  const int batch = blockIdx.z;
  src += (size_t)batch * R * C;
  dst += (size_t)batch * R * C;
  const int r0 = blockIdx.y * 32, c0 = blockIdx.x * 32;
  const int tx = threadIdx.x & 31, ty = threadIdx.x >> 5;
#pragma unroll
  for (int i = 0; i < 32; i += 8)
    t[ty + i][tx] = src[(size_t)(r0 + ty + i) * C + c0 + tx];
  __syncthreads();
#pragma unroll
  for (int i = 0; i < 32; i += 8)
    dst[(size_t)(c0 + ty + i) * R + r0 + tx] = (_Float16)t[tx][ty + i];
}

// ---------------- kernel 1: fused QKV projection ----------------
// grid (S/128, H, B), block 256 (8 waves). Weights arrive pre-transposed in a
// SINGLE buffer wt[3][H][KD][D] (q=0,k=1,v=2) so all staging addresses are
// plain arithmetic on one kernel-arg pointer -> global_load_b128 (no flat).
// V is written transposed ([B,H,KD,S]) with vector stores.
__global__ __launch_bounds__(256) void mha_qkv_kernel(
    const _Float16* __restrict__ xh,      // [B*S, D]
    const _Float16* __restrict__ wt,      // [3, H, KD, D]
    _Float16* __restrict__ q,             // [B, H, S, KD]
    _Float16* __restrict__ k,             // [B, H, S, KD]
    _Float16* __restrict__ vt) {          // [B, H, KD, S]
  __shared__ _Float16 lx[128 * 32];       // x tile, row-major [m][k]
  __shared__ _Float16 lwT[192 * 32];      // Wq|Wk|Wv tile, [n][k]
  const int s0   = blockIdx.x * 128;
  const int h    = blockIdx.y;
  const int b    = blockIdx.z;
  const int tid  = threadIdx.x;
  const int wave = tid >> 5;
  const int lane = tid & 31;

  v8f cq[4] = {}, ck[4] = {}, cv[4] = {};
  const _Float16* xrow = xh + (size_t)(b * S_ + s0) * D_;

  for (int k0 = 0; k0 < D_; k0 += 32) {
    {  // x tile 128x32: 2 threads per row, 2 x b128 each
      int r  = tid >> 1;
      int c0 = (tid & 1) << 4;
      const v8h* g = (const v8h*)(xrow + (size_t)r * D_ + k0 + c0);
      v8h* s = (v8h*)(lx + r * 32 + c0);
      s[0] = g[0]; s[1] = g[1];
    }
#pragma unroll
    for (int i = 0; i < 3; ++i) {  // W^T tile 192x32: 768 b128 chunks
      int chunk = tid + i * 256;
      int row = chunk >> 2;                // 0..191 (n index: wq|wk|wv x kd)
      int c0  = (chunk & 3) << 3;
      int wsl = row >> 6;
      int kd  = row & 63;
      *(v8h*)(lwT + row * 32 + c0) =
          *(const v8h*)(wt + ((size_t)(wsl * H_ + h) * KD_ + kd) * D_ + k0 + c0);
    }
    __syncthreads();
    v16h a = load_a16x32(lx + wave * 16 * 32, 32);
#pragma unroll
    for (int j = 0; j < 4; ++j) {
      v16h bq = load_bT(lwT + (0   + j * 16) * 32, 32);
      v16h bk = load_bT(lwT + (64  + j * 16) * 32, 32);
      v16h bv = load_bT(lwT + (128 + j * 16) * 32, 32);
      cq[j] = wmma16(a, bq, cq[j]);
      ck[j] = wmma16(a, bk, ck[j]);
      cv[j] = wmma16(a, bv, cv[j]);
    }
    __syncthreads();
  }
  const int n  = lane & 15;
  const int mb = (lane >> 4) << 3;
  const int srow = s0 + wave * 16 + mb;
  size_t qkbase = ((size_t)(b * H_ + h) * S_ + srow) * KD_;
#pragma unroll
  for (int j = 0; j < 4; ++j)
#pragma unroll
    for (int r = 0; r < 8; ++r) {
      q[qkbase + (size_t)r * KD_ + j * 16 + n] = (_Float16)cq[j][r];
      k[qkbase + (size_t)r * KD_ + j * 16 + n] = (_Float16)ck[j][r];
    }
  // V transposed: lane's 8 rows are consecutive s -> contiguous b128 store
  size_t vtbase = (size_t)(b * H_ + h) * KD_;
#pragma unroll
  for (int j = 0; j < 4; ++j) {
    v8h pack;
#pragma unroll
    for (int r = 0; r < 8; ++r) pack[r] = (_Float16)cv[j][r];
    *(v8h*)(vt + (vtbase + j * 16 + n) * S_ + srow) = pack;
  }
}

// ---------------- kernel 2: flash attention ----------------
// grid (S/128, H, B), block 256. K tile row-major [sk][kd] (== col-major K^T),
// V tile pre-transposed [kd][sk]; all fragment loads contiguous.
__global__ __launch_bounds__(256) void mha_attn_kernel(
    const _Float16* __restrict__ q,       // [B,H,S,KD]
    const _Float16* __restrict__ kk,      // [B,H,S,KD]
    const _Float16* __restrict__ vt,      // [B,H,KD,S]
    _Float16* __restrict__ z) {           // [B, S, H*KD]
  __shared__ _Float16 lK[64 * 64];        // [sk][kd]
  __shared__ _Float16 lVt[64 * 64];       // [kd][sk]
  __shared__ _Float16 lp[8 * 16 * 64];    // per-wave P staging, [m][sk]
  const int h    = blockIdx.y;
  const int b    = blockIdx.z;
  const int tid  = threadIdx.x;
  const int wave = tid >> 5;
  const int lane = tid & 31;
  const int qs0  = blockIdx.x * 128 + wave * 16;
  const size_t bh   = (size_t)(b * H_ + h) * S_;
  const size_t bhKD = (size_t)(b * H_ + h) * KD_;

  v16h qa0 = load_a16x32(q + (bh + qs0) * KD_, KD_);
  v16h qa1 = load_a16x32(q + (bh + qs0) * KD_ + 32, KD_);

  v8f o[4] = {};
  float mrow[8], lrow[8];
#pragma unroll
  for (int r = 0; r < 8; ++r) { mrow[r] = -1e30f; lrow[r] = 0.0f; }

  _Float16* myp = lp + wave * (16 * 64);
  const float scale = 0.125f;  // 1/sqrt(KD)

  for (int t0 = 0; t0 < S_; t0 += 64) {
    {  // cooperative tile loads: 512 b128 chunks each
      const _Float16* kg = kk + (bh + t0) * KD_;
      const _Float16* vg = vt + bhKD * S_ + t0;
#pragma unroll
      for (int i = 0; i < 2; ++i) {
        int chunk = tid + i * 256;
        int r  = chunk >> 3;
        int c0 = (chunk & 7) << 3;
        *(v8h*)(lK  + r * 64 + c0) = *(const v8h*)(kg + (size_t)r * KD_ + c0);
        *(v8h*)(lVt + r * 64 + c0) = *(const v8h*)(vg + (size_t)r * S_  + c0);
      }
    }
    __syncthreads();

    // scores = Q * K^T  (16x64 tile per wave)
    v8f sc[4];
#pragma unroll
    for (int j = 0; j < 4; ++j) {
      v8f zero = {};
      v16h b0 = load_bT(lK + (j * 16) * 64, 64);        // kd 0..31
      v16h b1 = load_bT(lK + (j * 16) * 64 + 32, 64);   // kd 32..63
      sc[j] = wmma16(qa1, b1, wmma16(qa0, b0, zero));
    }
#pragma unroll
    for (int j = 0; j < 4; ++j)
#pragma unroll
      for (int r = 0; r < 8; ++r) sc[j][r] *= scale;

    // online softmax: row max over 4 frags + 16-lane butterfly
    float mnew[8], alpha[8];
#pragma unroll
    for (int r = 0; r < 8; ++r) {
      float m = fmaxf(fmaxf(sc[0][r], sc[1][r]), fmaxf(sc[2][r], sc[3][r]));
#pragma unroll
      for (int d = 1; d < 16; d <<= 1) m = fmaxf(m, __shfl_xor(m, d, 32));
      mnew[r]  = fmaxf(mrow[r], m);
      alpha[r] = __expf(mrow[r] - mnew[r]);
      mrow[r]  = mnew[r];
    }
    float lad[8] = {};
#pragma unroll
    for (int j = 0; j < 4; ++j)
#pragma unroll
      for (int r = 0; r < 8; ++r) {
        float p = __expf(sc[j][r] - mnew[r]);
        sc[j][r] = p;
        lad[r] += p;
      }
#pragma unroll
    for (int r = 0; r < 8; ++r) {
      float s = lad[r];
#pragma unroll
      for (int d = 1; d < 16; d <<= 1) s += __shfl_xor(s, d, 32);
      lrow[r] = lrow[r] * alpha[r] + s;
    }
#pragma unroll
    for (int j = 0; j < 4; ++j)
#pragma unroll
      for (int r = 0; r < 8; ++r) o[j][r] *= alpha[r];

    // C-layout -> A-layout through wave-private LDS (DS ops in-order per wave)
    {
      const int n  = lane & 15;
      const int mb = (lane >> 4) << 3;
#pragma unroll
      for (int j = 0; j < 4; ++j)
#pragma unroll
        for (int r = 0; r < 8; ++r)
          myp[(mb + r) * 64 + j * 16 + n] = (_Float16)sc[j][r];
    }
    v16h pa0 = load_a16x32(myp, 64);
    v16h pa1 = load_a16x32(myp + 32, 64);
#pragma unroll
    for (int j = 0; j < 4; ++j) {
      v16h b0 = load_bT(lVt + (j * 16) * 64, 64);       // sk 0..31
      v16h b1 = load_bT(lVt + (j * 16) * 64 + 32, 64);  // sk 32..63
      o[j] = wmma16(pa1, b1, wmma16(pa0, b0, o[j]));
    }
    __syncthreads();
  }

  {  // epilogue: normalize + store z[b, s, h*KD + kd]
    const int n  = lane & 15;
    const int mb = (lane >> 4) << 3;
    float inv[8];
#pragma unroll
    for (int r = 0; r < 8; ++r) inv[r] = 1.0f / lrow[r];
    _Float16* zo = z + ((size_t)(b * S_) + qs0 + mb) * (H_ * KD_) + h * KD_;
#pragma unroll
    for (int j = 0; j < 4; ++j)
#pragma unroll
      for (int r = 0; r < 8; ++r)
        zo[(size_t)r * (H_ * KD_) + j * 16 + n] = (_Float16)(o[j][r] * inv[r]);
  }
}

// ---------------- kernel 3: output projection ----------------
// out[M=B*S, N=D] = z[M, 640] @ Wo[640, D]; Wo arrives transposed [D][F].
__global__ __launch_bounds__(256) void mha_proj_kernel(
    const _Float16* __restrict__ zh,      // [B*S, 640]
    const _Float16* __restrict__ wot,     // [D, F] = Wo^T
    float* __restrict__ out) {            // [B*S, 640]
  __shared__ _Float16 lz[128 * 32];       // [m][k]
  __shared__ _Float16 lwT[64 * 32];       // [n][k]
  const int m0   = blockIdx.x * 128;
  const int n0   = blockIdx.y * 64;
  const int tid  = threadIdx.x;
  const int wave = tid >> 5;
  const int lane = tid & 31;
  v8f acc[4] = {};

  for (int k0 = 0; k0 < D_; k0 += 32) {
    {
      int r  = tid >> 1;
      int c0 = (tid & 1) << 4;
      const v8h* g = (const v8h*)(zh + (size_t)(m0 + r) * D_ + k0 + c0);
      v8h* s = (v8h*)(lz + r * 32 + c0);
      s[0] = g[0]; s[1] = g[1];
    }
    {  // Wo^T tile 64x32: 256 b128 chunks, one per thread
      int row = tid >> 2;
      int c0  = (tid & 3) << 3;
      *(v8h*)(lwT + row * 32 + c0) =
          *(const v8h*)(wot + (size_t)(n0 + row) * D_ + k0 + c0);
    }
    __syncthreads();
    v16h a = load_a16x32(lz + wave * 16 * 32, 32);
#pragma unroll
    for (int j = 0; j < 4; ++j) {
      v16h bb = load_bT(lwT + (j * 16) * 32, 32);
      acc[j] = wmma16(a, bb, acc[j]);
    }
    __syncthreads();
  }
  const int n  = lane & 15;
  const int mb = (lane >> 4) << 3;
  float* po = out + (size_t)(m0 + wave * 16 + mb) * D_ + n0;
#pragma unroll
  for (int j = 0; j < 4; ++j)
#pragma unroll
    for (int r = 0; r < 8; ++r)
      po[(size_t)r * D_ + j * 16 + n] = acc[j][r];
}

// ---------------- launcher ----------------
extern "C" void kernel_launch(void* const* d_in, const int* in_sizes, int n_in,
                              void* d_out, int out_size, void* d_ws, size_t ws_size,
                              hipStream_t stream) {
  (void)in_sizes; (void)n_in; (void)out_size; (void)ws_size;
  const float* x  = (const float*)d_in[0];
  const float* Wq = (const float*)d_in[1];
  const float* Wk = (const float*)d_in[2];
  const float* Wv = (const float*)d_in[3];
  const float* Wo = (const float*)d_in[4];
  float* out = (float*)d_out;

  size_t off = 0;
  char* wsb = (char*)d_ws;
  auto alloc = [&](size_t bytes) -> void* {
    void* p = wsb + off;
    off += (bytes + 255) & ~(size_t)255;
    return p;
  };
  const size_t nx  = (size_t)B_ * S_ * D_;
  const size_t nw  = (size_t)H_ * D_ * KD_;   // per projection
  const size_t nwo = (size_t)D_ * D_;
  const size_t nqk = (size_t)B_ * H_ * S_ * KD_;

  _Float16* xh  = (_Float16*)alloc(nx * 2);
  _Float16* wt  = (_Float16*)alloc(3 * nw * 2);  // [3,H,KD,D] contiguous
  _Float16* wot = (_Float16*)alloc(nwo * 2);     // [D,F]
  _Float16* qb  = (_Float16*)alloc(nqk * 2);     // [B,H,S,KD]
  _Float16* kb  = (_Float16*)alloc(nqk * 2);     // [B,H,S,KD]
  _Float16* vtb = (_Float16*)alloc(nqk * 2);     // [B,H,KD,S]
  _Float16* zb  = (_Float16*)alloc(nx * 2);      // [B,S,H*KD]

  mha_cvt_kernel<<<(int)((nx + 255) / 256), 256, 0, stream>>>(x, xh, (int)nx);
  // Wq/Wk/Wv: per-head transpose [D][KD] -> [KD][D], into slices of wt
  mha_cvt_t_kernel<<<dim3(KD_ / 32, D_ / 32, H_), 256, 0, stream>>>(
      Wq, wt + 0 * nw, D_, KD_);
  mha_cvt_t_kernel<<<dim3(KD_ / 32, D_ / 32, H_), 256, 0, stream>>>(
      Wk, wt + 1 * nw, D_, KD_);
  mha_cvt_t_kernel<<<dim3(KD_ / 32, D_ / 32, H_), 256, 0, stream>>>(
      Wv, wt + 2 * nw, D_, KD_);
  // Wo: [F][D] -> [D][F]
  mha_cvt_t_kernel<<<dim3(D_ / 32, D_ / 32, 1), 256, 0, stream>>>(Wo, wot, D_, D_);

  mha_qkv_kernel<<<dim3(S_ / 128, H_, B_), 256, 0, stream>>>(
      xh, wt, qb, kb, vtb);

  mha_attn_kernel<<<dim3(S_ / 128, H_, B_), 256, 0, stream>>>(qb, kb, vtb, zb);

  mha_proj_kernel<<<dim3((B_ * S_) / 128, D_ / 64), 256, 0, stream>>>(
      zb, wot, out);
}
